// LongShortAggregator_28561532518858
// MI455X (gfx1250) — matrix operand order repaired and checked by reading, not compiled, not run on hardware
//
#include <hip/hip_runtime.h>
#include <hip/hip_bf16.h>
#include <math.h>

typedef __attribute__((ext_vector_type(16))) _Float16 v16h;
typedef __attribute__((ext_vector_type(8)))  float    v8f;

#define NTHREADS 256
#define NWAVES 8

// ---------------- WMMA fragment helpers (wave32; A/B in contiguous k order, C/D per CDNA5 ISA 7.12.2) ----------------

// A operand: 16x32 f16 tile from row-major f16 (src = tile base, leading dim ld).
// Element e: K = e + (lane>=16 ? 16 : 0); 32 contiguous bytes per lane.
__device__ __forceinline__ v16h frag_a(const _Float16* __restrict__ src, int ld) {
  int lane = threadIdx.x & 31;
  int m = lane & 15;
  int kb = (lane & 16) ? 16 : 0;
  v16h a;
#pragma unroll
  for (int e = 0; e < 16; ++e) {
    int k = e + kb;
    a[e] = src[m * ld + k];
  }
  return a;
}

// B operand for C = X @ W^T with W stored [n][k] (row-major N_out x K), src = W + n0*ld + k0.
// Lane 0-15: N=lane, K base 0; lane 16-31: N=lane-16, K base 16. 32 contiguous bytes per lane.
__device__ __forceinline__ v16h frag_bT(const _Float16* __restrict__ src, int ld) {
  int lane = threadIdx.x & 31;
  int n = lane & 15;
  int kb = (lane & 16) ? 16 : 0;
  v16h b;
#pragma unroll
  for (int e = 0; e < 16; ++e) b[e] = src[n * ld + kb + e];
  return b;
}

// C/D 32-bit 16x16: VGPR r: lanes 0-15 -> (M=r, N=lane), lanes 16-31 -> (M=r+8, N=lane-16).
__device__ __forceinline__ v8f frag_c_load(const float* __restrict__ src, int ld) {
  int lane = threadIdx.x & 31;
  int n = lane & 15;
  int mb = (lane & 16) ? 8 : 0;
  v8f c;
#pragma unroll
  for (int r = 0; r < 8; ++r) c[r] = src[(mb + r) * ld + n];
  return c;
}

__device__ __forceinline__ v8f wmma_f16(v16h a, v16h b, v8f c) {
  // (neg_a, A, neg_b, B, c_mod, C, reuse_a, reuse_b)
  return __builtin_amdgcn_wmma_f32_16x16x32_f16(false, a, false, b, (short)0, c,
                                                false, false);
}

// C(64x128) = A(64xK) @ B^T(Kx128) [+bias] [+C if ACC].
// Wave w owns the column stripe n0 = 16*w for ALL row blocks m0 = 0,16,32,48:
//   -> one shared B fragment (and bias value) per k-step feeds 4 independent
//      accumulator chains (hides WMMA hazards, 4x fewer B loads, no spills).
// A: f16 LDS row-major (lda). B: f16 "weight layout" [n][k] with row stride ldb.
// WCS: write f32 Cs (ld 128). WCH: write f16 Ch; CHT -> transposed [n][m] with ld ldch.
template<bool ACC, bool HASBIAS, bool WCS, bool WCH, bool CHT>
__device__ __forceinline__ void mm64x128(
    const _Float16* __restrict__ As, int lda, int K,
    const _Float16* __restrict__ Bs, int ldb,
    const float* __restrict__ bias,
    float* __restrict__ Cs,
    _Float16* __restrict__ Ch, int ldch)
{
  const int wave = threadIdx.x >> 5;
  const int lane = threadIdx.x & 31;
  const int n_in = lane & 15;
  const int m_in = (lane & 16) ? 8 : 0;
  const int n0 = wave << 4;

  v8f c[4];
#pragma unroll
  for (int tt = 0; tt < 4; ++tt) {
    if (ACC) c[tt] = frag_c_load(Cs + (tt << 4) * 128 + n0, 128);
    else     c[tt] = (v8f){};
  }
  for (int k0 = 0; k0 < K; k0 += 32) {
    v16h bfr = frag_bT(Bs + n0 * ldb + k0, ldb);   // shared across the 4 row tiles
#pragma unroll
    for (int tt = 0; tt < 4; ++tt) {
      v16h a = frag_a(As + (tt << 4) * lda + k0, lda);
      c[tt] = wmma_f16(a, bfr, c[tt]);
    }
  }
  const float bv = HASBIAS ? bias[n0 + n_in] : 0.0f;
#pragma unroll
  for (int tt = 0; tt < 4; ++tt) {
    const int row = (tt << 4) + m_in;
    const int col = n0 + n_in;
#pragma unroll
    for (int r = 0; r < 8; ++r) {
      float v = c[tt][r] + bv;
      if (WCS) Cs[(row + r) * 128 + col] = v;
      if (WCH) {
        if (CHT) Ch[col * ldch + row + r] = (_Float16)v;   // contiguous: 1x b128 store
        else     Ch[(row + r) * 128 + col] = (_Float16)v;
      }
    }
  }
}

// Async global -> LDS copy of 16 bytes per lane (GLOBAL_LOAD_ASYNC_TO_LDS_B128, ASYNCcnt).
// LDS address = low 32 bits of the generic pointer (ISA: LDS aperture keeps offset in [31:0]).
__device__ __forceinline__ void async_copy_b128(const float* gsrc, float* ldst) {
  unsigned l = (unsigned)(unsigned long long)ldst;
  asm volatile("global_load_async_to_lds_b128 %0, %1, off"
               :: "v"(l), "v"(gsrc) : "memory");
}
__device__ __forceinline__ void async_wait0() {
  asm volatile("s_wait_asynccnt 0x0" ::: "memory");
}

// ---------------- weight f32 -> f16 conversion ----------------
__global__ void cvt_f16_kernel(const float* __restrict__ src,
                               _Float16* __restrict__ dst, int n) {
  int i = blockIdx.x * blockDim.x + threadIdx.x;
  if (i < n) dst[i] = (_Float16)src[i];
}

// ---------------- fused per-batch pipeline ----------------
// LDS layout (bytes):
//   0      sH   f32 64x128   (current hidden / sate)
//   32768  sHh  f16 64x128
//   49152  sAh  f16 64x128   (A, both halves)
//   65536  sTh  f16          (hvT 128x64 transposed / orig-hidden f16 row-major)
//   81920  sInh f16 64x128   (inp_in)     } aliased by sNn f32 64x128 (h_n)
//   98304  sOuth f16 64x128  (inp_out)    }
//   114688 sZ   f32 64x128   (also async staging for A at start)
//   147456 sR   f32 64x128   (r, then hs)
//   180224 sT   f32 64x128   (i_n / q_items / k_items / orig hidden)
//   212992 sVec1, 213504 sVec2, 214016 sStar, 214528 sAw, 214784 sPart, 215808 sRed
#define SMEM_TOTAL 216064

__global__ __launch_bounds__(NTHREADS, 1) void sgnn_kernel(
    const float* __restrict__ gA,        // (B,64,128)
    const float* __restrict__ gHidden,   // (B,64,128)
    const float* __restrict__ gStar0,    // (B,128)
    const int*   __restrict__ gMask,     // (B,64)
    const float* __restrict__ bi,        // 384
    const float* __restrict__ bh,        // 384
    const float* __restrict__ b_in,  const float* __restrict__ b_out,
    const float* __restrict__ b_iah, const float* __restrict__ b_oah,
    const float* __restrict__ Wk1_f32,   // 128x128 (f32, small ops)
    const float* __restrict__ Wq2_f32,   // 128x128
    const float* __restrict__ bw,        // 128
    const _Float16* __restrict__ h_wi,   // 384x256
    const _Float16* __restrict__ h_wh,   // 384x128
    const _Float16* __restrict__ h_Win,  // 128x128
    const _Float16* __restrict__ h_Wout, // 128x128
    const _Float16* __restrict__ h_Wq1,  // 128x128
    const _Float16* __restrict__ h_Wk2,  // 128x128
    const _Float16* __restrict__ h_Ww,   // 128x256
    float* __restrict__ gOutH,           // (B,64,128)
    float* __restrict__ gOutS)           // (B,128)
{
  __shared__ __align__(16) unsigned char smem[SMEM_TOTAL];
  float*    sH    = (float*)(smem + 0);
  _Float16* sHh   = (_Float16*)(smem + 32768);
  _Float16* sAh   = (_Float16*)(smem + 49152);
  _Float16* sTh   = (_Float16*)(smem + 65536);
  _Float16* sInh  = (_Float16*)(smem + 81920);
  _Float16* sOuth = (_Float16*)(smem + 98304);
  float*    sNn   = (float*)(smem + 81920);   // aliases sInh+sOuth (h_n)
  float*    sZ    = (float*)(smem + 114688);
  float*    sR    = (float*)(smem + 147456);
  float*    sT    = (float*)(smem + 180224);
  float*    sVec1 = (float*)(smem + 212992);
  float*    sVec2 = (float*)(smem + 213504);
  float*    sStar = (float*)(smem + 214016);
  float*    sAw   = (float*)(smem + 214528);
  float*    sPart = (float*)(smem + 214784);
  float*    sRed  = (float*)(smem + 215808);

  const int tid = threadIdx.x;
  const int b = blockIdx.x;
  const float scale = 0.0883883476483184f; // 1/sqrt(128)

  if (tid == 0) {
    __builtin_prefetch(h_wi, 0, 0);
    __builtin_prefetch(h_wh, 0, 0);
    __builtin_prefetch(h_Win, 0, 0);
  }

  const float* gHb = gHidden + (size_t)b * 8192;
  const float* gAb = gA + (size_t)b * 8192;

  // Async DMA: hidden -> sH, A -> sZ (staging), 16B per lane per issue.
  for (int i = tid * 4; i < 8192; i += NTHREADS * 4) {
    async_copy_b128(gHb + i, sH + i);
    async_copy_b128(gAb + i, sZ + i);
  }
  if (tid < 128) sStar[tid] = gStar0[(size_t)b * 128 + tid];
  async_wait0();
  __syncthreads();
  for (int i = tid; i < 8192; i += NTHREADS) {
    sHh[i] = (_Float16)sH[i];
    sAh[i] = (_Float16)sZ[i];
  }
  __syncthreads();

  for (int step = 0; step < 2; ++step) {
    // ================= GGNN =================
    // hv_in = h @ W_in^T + b_in  -> transposed f16 hvT (128x64)
    mm64x128<false, true, false, true, true>(sHh, 128, 128, h_Win, 128, b_in, nullptr, sTh, 64);
    __syncthreads();   // sTh producer -> consumer
    // inp_in = A_in @ hv_in + b_iah (hvT acts as weight-layout B)
    mm64x128<false, true, false, true, false>(sAh, 128, 64, sTh, 64, b_iah, nullptr, sInh, 128);
    __syncthreads();   // WAR on sTh (rewritten next)
    // hv_out = h @ W_out^T + b_out -> hvT
    mm64x128<false, true, false, true, true>(sHh, 128, 128, h_Wout, 128, b_out, nullptr, sTh, 64);
    __syncthreads();   // sTh producer -> consumer
    // inp_out = A_out @ hv_out + b_oah
    mm64x128<false, true, false, true, false>(sAh + 64, 128, 64, sTh, 64, b_oah, nullptr, sOuth, 128);
    __syncthreads();   // sInh/sOuth producers -> gate-block consumers
    // ---- gate block: 8 GEMMs, wave-private C tiles, read-only shared inputs.
    //      No internal barriers needed (intra-wave LDS RAW handled by dscnt waits).
    // z pre-activation (gate group 1)
    mm64x128<false, true, true, false, false>(sInh, 128, 128, h_wi + 128 * 256, 256, bi + 128, sZ, nullptr, 0);
    mm64x128<true, false, true, false, false>(sOuth, 128, 128, h_wi + 128 * 256 + 128, 256, nullptr, sZ, nullptr, 0);
    mm64x128<true, true, true, false, false>(sHh, 128, 128, h_wh + 128 * 128, 128, bh + 128, sZ, nullptr, 0);
    // r pre-activation (gate group 0)
    mm64x128<false, true, true, false, false>(sInh, 128, 128, h_wi, 256, bi, sR, nullptr, 0);
    mm64x128<true, false, true, false, false>(sOuth, 128, 128, h_wi + 128, 256, nullptr, sR, nullptr, 0);
    mm64x128<true, true, true, false, false>(sHh, 128, 128, h_wh, 128, bh, sR, nullptr, 0);
    // i_n (gate group 2) -> sT
    mm64x128<false, true, true, false, false>(sInh, 128, 128, h_wi + 256 * 256, 256, bi + 256, sT, nullptr, 0);
    mm64x128<true, false, true, false, false>(sOuth, 128, 128, h_wi + 256 * 256 + 128, 256, nullptr, sT, nullptr, 0);
    __syncthreads();   // WAR: h_n overwrites sNn (= sInh/sOuth region) still being read above
    // h_n -> sNn
    mm64x128<false, true, true, false, false>(sHh, 128, 128, h_wh + 256 * 128, 128, bh + 256, sNn, nullptr, 0);
    __syncthreads();   // gate GEMMs -> cross-wave elementwise
    // elementwise GRU combine: hs -> sR (in place over r)
    for (int i = tid; i < 8192; i += NTHREADS) {
      float z = 1.0f / (1.0f + __expf(-sZ[i]));
      float r = 1.0f / (1.0f + __expf(-sR[i]));
      float nv = tanhf(r * sNn[i] + sT[i]);
      sR[i] = (1.0f - z) * sH[i] + z * nv;
    }
    __syncthreads();
    // per-row L2 norm of hs (4 threads per row)
    {
      int row = tid >> 2, part = tid & 3;
      const float* pr = sR + row * 128 + part * 32;
      float s = 0.0f;
#pragma unroll
      for (int d = 0; d < 32; ++d) { float v = pr[d]; s += v * v; }
      sPart[tid] = s;
    }
    __syncthreads();
    if (tid < 64) {
      float s = sPart[4 * tid] + sPart[4 * tid + 1] + sPart[4 * tid + 2] + sPart[4 * tid + 3];
      sRed[tid] = 1.0f / fmaxf(sqrtf(s), 1e-12f);
    }
    __syncthreads();
    for (int i = tid; i < 8192; i += NTHREADS) {
      float v = sH[i] + sR[i] * sRed[i >> 7];
      sH[i] = v;
      sHh[i] = (_Float16)v;
    }
    __syncthreads();

    // ================= PGAT =================
    // q_items = items @ Wq1^T -> sT
    mm64x128<false, false, true, false, false>(sHh, 128, 128, h_Wq1, 128, nullptr, sT, nullptr, 0);
    // k_star = star @ Wk1^T / q_star = star @ Wq2^T (128 dots of 128 each)
    if (tid < 128) {
      float a = 0.0f;
      const float* wr = Wk1_f32 + tid * 128;
      for (int k = 0; k < 128; ++k) a += sStar[k] * wr[k];
      sVec1[tid] = a;
    } else {
      int d = tid - 128;
      float a = 0.0f;
      const float* wr = Wq2_f32 + d * 128;
      for (int k = 0; k < 128; ++k) a += sStar[k] * wr[k];
      sVec2[d] = a;
    }
    __syncthreads();   // q_items + k_star -> aw
    // aw[n] = (q_items[n] . k_star) * scale
    if (tid < 64) {
      float a = 0.0f;
      const float* qr = sT + tid * 128;
      for (int d = 0; d < 128; ++d) a += qr[d] * sVec1[d];
      sAw[tid] = a * scale;
    }
    __syncthreads();
    // sate = (1-aw)*items + aw*star  (in place over sH/sHh)
    for (int i = tid; i < 8192; i += NTHREADS) {
      float a = sAw[i >> 7];
      float v = (1.0f - a) * sH[i] + a * sStar[i & 127];
      sH[i] = v;
      sHh[i] = (_Float16)v;
    }
    __syncthreads();
    // k_items = sate @ Wk2^T -> sT
    mm64x128<false, false, true, false, false>(sHh, 128, 128, h_Wk2, 128, nullptr, sT, nullptr, 0);
    __syncthreads();   // k_items -> logits
    // logits + mask
    if (tid < 64) {
      float a = 0.0f;
      const float* kr = sT + tid * 128;
      for (int d = 0; d < 128; ++d) a += sVec2[d] * kr[d];
      a *= scale;
      if (gMask[(size_t)b * 64 + tid] == 0) a = -1e30f;
      sAw[tid] = a;
    }
    __syncthreads();
    if (tid == 0) {
      float m = -1e30f;
      for (int n = 0; n < 64; ++n) m = fmaxf(m, sAw[n]);
      sRed[0] = m;
    }
    __syncthreads();
    if (tid < 64) sAw[tid] = __expf(sAw[tid] - sRed[0]);
    __syncthreads();
    if (tid == 0) {
      float s = 0.0f;
      for (int n = 0; n < 64; ++n) s += sAw[n];
      sRed[1] = 1.0f / s;
    }
    __syncthreads();
    // star = softmax(logits) @ sate
    if (tid < 128) {
      float a = 0.0f;
      for (int n = 0; n < 64; ++n) a += sAw[n] * sH[n * 128 + tid];
      sStar[tid] = a * sRed[1];
    }
    __syncthreads();
  }

  // ================= final gate =================
  // reload original hidden via async DMA (f32 into sT), then f16 row-major into sTh
  for (int i = tid * 4; i < 8192; i += NTHREADS * 4) {
    async_copy_b128(gHb + i, sT + i);
  }
  async_wait0();
  __syncthreads();
  for (int i = tid; i < 8192; i += NTHREADS) sTh[i] = (_Float16)sT[i];
  __syncthreads();
  // w_pre = hidden @ Ww[:, :128]^T + sate @ Ww[:, 128:]^T + bw -> sZ
  // (accumulate chain: same wave-private tiles, no barrier between the two GEMMs)
  mm64x128<false, false, true, false, false>(sTh, 128, 128, h_Ww, 256, nullptr, sZ, nullptr, 0);
  mm64x128<true, true, true, false, false>(sHh, 128, 128, h_Ww + 128, 256, bw, sZ, nullptr, 0);
  __syncthreads();
  float* outHb = gOutH + (size_t)b * 8192;
  for (int i = tid; i < 8192; i += NTHREADS) {
    float w = 1.0f / (1.0f + __expf(-sZ[i]));
    outHb[i] = w * sT[i] + (1.0f - w) * sH[i];
  }
  if (tid < 128) gOutS[(size_t)b * 128 + tid] = sStar[tid];
}

// ---------------- host launch ----------------
extern "C" void kernel_launch(void* const* d_in, const int* in_sizes, int n_in,
                              void* d_out, int out_size, void* d_ws, size_t ws_size,
                              hipStream_t stream) {
  (void)n_in; (void)out_size; (void)ws_size;
  const float* A      = (const float*)d_in[0];
  // d_in[1] = adj (unused by reference)
  const float* hidden = (const float*)d_in[2];
  const float* s_node = (const float*)d_in[3];
  const int*   mask   = (const int*)d_in[4];
  const float* wi     = (const float*)d_in[5];
  const float* bi     = (const float*)d_in[6];
  const float* wh     = (const float*)d_in[7];
  const float* bh     = (const float*)d_in[8];
  const float* W_in   = (const float*)d_in[9];
  const float* b_in   = (const float*)d_in[10];
  const float* W_out  = (const float*)d_in[11];
  const float* b_out  = (const float*)d_in[12];
  const float* b_iah  = (const float*)d_in[13];
  const float* b_oah  = (const float*)d_in[14];
  const float* Wq1    = (const float*)d_in[15];
  const float* Wk1    = (const float*)d_in[16];
  const float* Wq2    = (const float*)d_in[17];
  const float* Wk2    = (const float*)d_in[18];
  const float* Ww     = (const float*)d_in[19];
  const float* bw     = (const float*)d_in[20];

  const int Btot = in_sizes[2] / (64 * 128);

  // f16 weight staging in workspace (~480 KB)
  _Float16* ws     = (_Float16*)d_ws;
  _Float16* h_wi   = ws;                 // 384*256 = 98304
  _Float16* h_wh   = h_wi + 98304;       // 384*128 = 49152
  _Float16* h_Win  = h_wh + 49152;       // 16384
  _Float16* h_Wout = h_Win + 16384;      // 16384
  _Float16* h_Wq1  = h_Wout + 16384;     // 16384
  _Float16* h_Wk2  = h_Wq1 + 16384;      // 16384
  _Float16* h_Ww   = h_Wk2 + 16384;      // 128*256 = 32768

  auto cvt = [&](const float* s, _Float16* d, int n) {
    cvt_f16_kernel<<<(n + 255) / 256, 256, 0, stream>>>(s, d, n);
  };
  cvt(wi, h_wi, 98304);
  cvt(wh, h_wh, 49152);
  cvt(W_in, h_Win, 16384);
  cvt(W_out, h_Wout, 16384);
  cvt(Wq1, h_Wq1, 16384);
  cvt(Wk2, h_Wk2, 16384);
  cvt(Ww, h_Ww, 32768);

  float* outH = (float*)d_out;
  float* outS = outH + (size_t)Btot * 8192;

  sgnn_kernel<<<Btot, NTHREADS, 0, stream>>>(
      A, hidden, s_node, mask, bi, bh, b_in, b_out, b_iah, b_oah,
      Wk1, Wq2, bw, h_wi, h_wh, h_Win, h_Wout, h_Wq1, h_Wk2, h_Ww,
      outH, outS);
}
